// JointNetwork_60327110640369
// MI455X (gfx1250) — compile-verified, matrix operand order checked by reading, not run on hardware
//
#include <hip/hip_runtime.h>

// Problem constants
#define BB 4
#define TT 256
#define UU 128
#define ED 512
#define JD 640
#define VV 1024

#define KT_COUNT 20          // JD / 32 k-steps
#define NTILES   64          // VV / 16 column tiles
#define JPITCH   648         // padded LDS pitch in halves (conflict-free banks)

typedef __attribute__((ext_vector_type(16))) _Float16 v16h;
typedef __attribute__((ext_vector_type(8)))  _Float16 h8;
typedef __attribute__((ext_vector_type(4)))  _Float16 h4;
typedef __attribute__((ext_vector_type(8)))  float    v8f;

// ---------------------------------------------------------------------------
// Kernel 1: enc_proj = enc @ W_enc^T + b_enc ; dec_proj = pred @ W_dec^T + b_dec
// ---------------------------------------------------------------------------
__global__ __launch_bounds__(256)
void proj_kernel(const float* __restrict__ enc, const float* __restrict__ dec,
                 const float* __restrict__ W_enc, const float* __restrict__ b_enc,
                 const float* __restrict__ W_dec, const float* __restrict__ b_dec,
                 float* __restrict__ encP, float* __restrict__ decP)
{
    __shared__ float sRow[ED];
    int blk = blockIdx.x;                      // 0..1023 enc rows, 1024..1535 dec rows
    const float* src; const float* W; const float* bias; float* dst;
    if (blk < BB * TT) {
        src = enc + (size_t)blk * ED;  W = W_enc; bias = b_enc;
        dst = encP + (size_t)blk * JD;
    } else {
        int r = blk - BB * TT;
        src = dec + (size_t)r * ED;    W = W_dec; bias = b_dec;
        dst = decP + (size_t)r * JD;
    }
    for (int i = threadIdx.x; i < ED; i += 256) sRow[i] = src[i];
    __syncthreads();

    const float4* s4 = (const float4*)sRow;
    for (int j = threadIdx.x; j < JD; j += 256) {
        const float4* w4 = (const float4*)(W + (size_t)j * ED);
        float acc = 0.f;
        #pragma unroll 4
        for (int k = 0; k < ED / 4; ++k) {
            float4 a = s4[k]; float4 w = w4[k];
            acc += a.x * w.x + a.y * w.y + a.z * w.z + a.w * w.w;
        }
        dst[j] = acc + bias[j];
    }
}

// ---------------------------------------------------------------------------
// Kernel 2: pack W_out (f32 [V][J]) into f16 B tiles, layout [kt][nt][lane][16h]
// so that within one k-step all 64 n-tiles are contiguous at 1KB stride.
// Per-lane WMMA 16-bit B layout: lane L -> col = L%16, laneHalf = L/16;
// half h: K = (h<8 ? 0 : 16) + laneHalf*8 + (h%8).
// ---------------------------------------------------------------------------
__global__ __launch_bounds__(256)
void pack_wout_kernel(const float* __restrict__ W_out, _Float16* __restrict__ Bp)
{
    int idx = blockIdx.x * 256 + threadIdx.x;      // KT_COUNT*NTILES*32*16 = 655360
    if (idx >= KT_COUNT * NTILES * 32 * 16) return;
    int h    = idx & 15;
    int lane = (idx >> 4) & 31;
    int nt   = (idx >> 9) & 63;
    int kt   = idx >> 15;
    int col  = nt * 16 + (lane & 15);
    int lh   = lane >> 4;
    int k    = kt * 32 + ((h < 8) ? (lh * 8 + h) : (16 + lh * 8 + (h - 8)));
    Bp[idx]  = (_Float16)W_out[(size_t)col * JD + k];
}

// ---------------------------------------------------------------------------
// Kernel 3: fused joint + output GEMM.
// Block = (b,t, 32-u tile), 512 threads = 16 waves.
// Phase 1: 32x640 tanh tile -> f16 LDS (vectorized x4).
// Phase 2: each wave owns N=64 (4 n-tiles) and BOTH 16-row M halves.
// Each B tile is loaded once per block and feeds two v_wmma_f32_16x16x32_f16.
// Small per-wave footprint (~130 VGPRs) -> full unroll, deep clauses, no
// VGPR-MSB switching. K=640 in 20 steps; next B slice prefetched each step.
// ---------------------------------------------------------------------------
__global__ __launch_bounds__(512)
void joint_kernel(const float* __restrict__ encP, const float* __restrict__ decP,
                  const _Float16* __restrict__ Bp, const float* __restrict__ b_out,
                  float* __restrict__ out)
{
    __shared__ _Float16 sJ[32 * JPITCH];

    int blk = blockIdx.x;            // 0..4095
    int ut  = blk & 3;               // u-tile within U
    int bt  = blk >> 2;              // b*T + t
    int b   = bt >> 8;               // T = 256
    int u0  = ut * 32;

    const float* eRow  = encP + (size_t)bt * JD;
    const float* dRow0 = decP + (size_t)(b * UU + u0) * JD;

    // Phase 1: tanh(enc_proj + dec_proj) tile -> f16 LDS, 4 elements per step
    for (int e = threadIdx.x; e < 32 * (JD / 4); e += 512) {
        int u  = e / (JD / 4);
        int j4 = e - u * (JD / 4);
        float4 ev = *(const float4*)(eRow + j4 * 4);
        float4 dv = *(const float4*)(dRow0 + (size_t)u * JD + j4 * 4);
        h4 hv;
        hv.x = (_Float16)tanhf(ev.x + dv.x);
        hv.y = (_Float16)tanhf(ev.y + dv.y);
        hv.z = (_Float16)tanhf(ev.z + dv.z);
        hv.w = (_Float16)tanhf(ev.w + dv.w);
        *(h4*)&sJ[u * JPITCH + j4 * 4] = hv;
    }
    __syncthreads();

    int lane = threadIdx.x & 31;
    int w    = threadIdx.x >> 5;     // wave id 0..15 -> N cols [w*64, w*64+64)
    int lh   = lane >> 4;            // lane half (K-half selector)
    int rA   = lane & 15;            // A row within 16-row tile

    v8f acc0[4] = {};                // M rows u0+0..15
    v8f acc1[4] = {};                // M rows u0+16..31

    // Per-wave B base: n-tile stride = 32 v16h (1KB), per-kt stride = 2048 (64KB)
    const v16h* bp = (const v16h*)Bp + w * 128 + lane;
    const _Float16* aRow0 = &sJ[rA * JPITCH + lh * 8];
    const _Float16* aRow1 = &sJ[(16 + rA) * JPITCH + lh * 8];

    #pragma unroll
    for (int kt = 0; kt < KT_COUNT; ++kt) {
        // A operands for both M halves: 16B LDS reads with immediate offsets
        h8 lo0 = *(const h8*)(aRow0 + kt * 32);
        h8 hi0 = *(const h8*)(aRow0 + kt * 32 + 16);
        h8 lo1 = *(const h8*)(aRow1 + kt * 32);
        h8 hi1 = *(const h8*)(aRow1 + kt * 32 + 16);
        v16h a0 = __builtin_shufflevector(lo0, hi0, 0,1,2,3,4,5,6,7,8,9,10,11,12,13,14,15);
        v16h a1 = __builtin_shufflevector(lo1, hi1, 0,1,2,3,4,5,6,7,8,9,10,11,12,13,14,15);

        // Prefetch next 4KB k-slice of this wave's B region (one line per lane)
        if (kt + 1 < KT_COUNT) {
            const char* pf = (const char*)(bp + 2048 - lane) + lane * 128;
            __builtin_prefetch(pf, 0, 1);
        }

        // Stage this wave's 4 B tiles (immediate offsets); each feeds two WMMAs
        v16h bv[4];
        #pragma unroll
        for (int nt = 0; nt < 4; ++nt)
            bv[nt] = bp[nt * 32];
        #pragma unroll
        for (int nt = 0; nt < 4; ++nt) {
            acc0[nt] = __builtin_amdgcn_wmma_f32_16x16x32_f16(
                false, a0, false, bv[nt], (short)0, acc0[nt], false, false);
            acc1[nt] = __builtin_amdgcn_wmma_f32_16x16x32_f16(
                false, a1, false, bv[nt], (short)0, acc1[nt], false, false);
        }
        bp += 2048;
    }

    // Store: C/D layout -> lane L: col = L%16, rows M = r + 8*(L>=16).
    // Single base; all bias loads and stores use immediate offsets.
    int colBase = w * 64 + rA;
    const float* bop = b_out + colBase;
    float* op0 = out + ((size_t)(bt * UU + u0 + lh * 8)) * VV + colBase;
    #pragma unroll
    for (int nt = 0; nt < 4; ++nt) {
        float bo = bop[nt * 16];
        #pragma unroll
        for (int r = 0; r < 8; ++r) {
            op0[nt * 16 + r * VV]        = acc0[nt][r] + bo;   // rows u0 + lh*8 + r
            op0[nt * 16 + (16 + r) * VV] = acc1[nt][r] + bo;   // rows +16
        }
    }
}

// ---------------------------------------------------------------------------
extern "C" void kernel_launch(void* const* d_in, const int* in_sizes, int n_in,
                              void* d_out, int out_size, void* d_ws, size_t ws_size,
                              hipStream_t stream) {
    const float* enc   = (const float*)d_in[0];
    const float* pred  = (const float*)d_in[1];
    const float* W_enc = (const float*)d_in[2];
    const float* b_enc = (const float*)d_in[3];
    const float* W_dec = (const float*)d_in[4];
    const float* b_dec = (const float*)d_in[5];
    const float* W_out = (const float*)d_in[6];
    const float* b_out = (const float*)d_in[7];
    float* out = (float*)d_out;

    // Workspace layout (f32): encP [1024*640], decP [512*640], then f16 Bpack
    float*    encP = (float*)d_ws;
    float*    decP = encP + (size_t)BB * TT * JD;
    _Float16* Bp   = (_Float16*)((char*)d_ws +
                       ((size_t)BB * TT * JD + (size_t)BB * UU * JD) * sizeof(float));

    proj_kernel<<<BB * TT + BB * UU, 256, 0, stream>>>(
        enc, pred, W_enc, b_enc, W_dec, b_dec, encP, decP);

    int packN = KT_COUNT * NTILES * 32 * 16;
    pack_wout_kernel<<<(packN + 255) / 256, 256, 0, stream>>>(W_out, Bp);

    joint_kernel<<<BB * TT * (UU / 32), 512, 0, stream>>>(encP, decP, Bp, b_out, out);
}
